// WeatherDecoder_12171937316942
// MI455X (gfx1250) — matrix operand order repaired and checked by reading, not compiled, not run on hardware
//
#include <hip/hip_runtime.h>
#include <hip/hip_bf16.h>

// ---------------- problem constants ----------------
#define D_    1024
#define NH_   16
#define DH_   64
#define P_    2
#define NLEV_ 7
#define NVA_  5
#define NVS_  4
#define HR_   64
#define WR_   128
#define HW_   (HR_ * WR_)     // 8192
#define H_    (HR_ * P_)      // 128
#define W_    (WR_ * P_)      // 256
#define ML_   (HW_ * NLEV_)   // 57344

typedef _Float16 half_t;
typedef _Float16 h16v __attribute__((ext_vector_type(16)));
typedef _Float16 h8v  __attribute__((ext_vector_type(8)));
typedef float    f8v  __attribute__((ext_vector_type(8)));

#define OUT_F32 1
#define OUT_F16 2

// ---------------------------------------------------------------------------
// WMMA GEMM:  C(MxN) = A(MxK,f16) @ Bw(NxK,f16)^T + bias(N)
// 256 threads = 8 waves arranged WMxWN; wave tile = (MT*16) x (NT*16).
// Block tile = (WM*MT*16) x (WN*NT*16).  K % 32 == 0.
// OUT: compile-time bitmask of which outputs to write (f32 and/or f16).
// ---------------------------------------------------------------------------
template<int MT, int NT, int WM, int WN, int OUT>
__global__ __launch_bounds__(256)
void k_gemm_wmma(const half_t* __restrict__ A, const half_t* __restrict__ Bw,
                 const float* __restrict__ bias,
                 float* __restrict__ Cf, half_t* __restrict__ Ch,
                 int N, int K) {
  const int lane = threadIdx.x & 31;
  const int wave = threadIdx.x >> 5;
  const int wm   = wave / WN;
  const int wn   = wave % WN;
  const int m0   = blockIdx.y * (WM * MT * 16) + wm * (MT * 16);
  const int n0   = blockIdx.x * (WN * NT * 16) + wn * (NT * 16);
  const int r15  = lane & 15;
  const int hi   = lane >> 4;

  // A fragment (16-bit A 16x32): lanes 0-15 hold K[0..7]+K[16..23] of row m,
  // lanes 16-31 hold K[8..15]+K[24..31] of the same row.
  const half_t* Ap = A  + (size_t)(m0 + r15) * K;
  // B fragment (32x16): lane n<16 holds K[0..7]+K[16..23] of column n0+n,
  // lane n+16 holds K[8..15]+K[24..31].  Column of B == row of Bw (N,K) row-major.
  const half_t* Bp = Bw + (size_t)(n0 + r15) * K;

  f8v acc[MT][NT] = {};

  for (int kb = 0; kb < K; kb += 32) {
    // speculative stream prefetch (global_prefetch_b8), 4 K-steps ahead
    #pragma unroll
    for (int i = 0; i < MT; ++i)
      __builtin_prefetch(Ap + (size_t)i * 16 * K + kb + 128, 0, 0);

    h16v a[MT], b[NT];
    #pragma unroll
    for (int i = 0; i < MT; ++i) {
      h8v lo = *(const h8v*)(Ap + (size_t)i * 16 * K + kb + hi * 8);
      h8v hh = *(const h8v*)(Ap + (size_t)i * 16 * K + kb + 16 + hi * 8);
      a[i] = __builtin_shufflevector(lo, hh, 0,1,2,3,4,5,6,7,8,9,10,11,12,13,14,15);
    }
    #pragma unroll
    for (int j = 0; j < NT; ++j) {
      h8v lo = *(const h8v*)(Bp + (size_t)j * 16 * K + kb + hi * 8);
      h8v hh = *(const h8v*)(Bp + (size_t)j * 16 * K + kb + 16 + hi * 8);
      b[j] = __builtin_shufflevector(lo, hh, 0,1,2,3,4,5,6,7,8,9,10,11,12,13,14,15);
    }

    #pragma unroll
    for (int i = 0; i < MT; ++i)
      #pragma unroll
      for (int j = 0; j < NT; ++j)
        acc[i][j] = __builtin_amdgcn_wmma_f32_16x16x32_f16(
            false, a[i], false, b[j], (short)0, acc[i][j], false, false);
  }

  // C layout: VGPR r, lanes 0-15 -> row r, lanes 16-31 -> row 8+r; col = lane&15.
  const int crow = hi * 8;
  #pragma unroll
  for (int i = 0; i < MT; ++i)
    #pragma unroll
    for (int j = 0; j < NT; ++j) {
      const f8v av = acc[i][j];
      const int mb = m0 + i * 16 + crow;
      const int nb = n0 + j * 16 + r15;
      const float bv = bias[nb];
      unsigned base = (unsigned)mb * (unsigned)N + (unsigned)nb;  // < 2^31
      #pragma unroll
      for (int r = 0; r < 8; ++r) {
        float v = av[r] + bv;
        unsigned idx = base + (unsigned)(r * N);
        if (OUT & OUT_F32) Cf[idx] = v;
        if (OUT & OUT_F16) Ch[idx] = (half_t)v;
      }
    }
}

// ---------------- glue kernels ----------------
__global__ void k_f32_to_f16(const float* __restrict__ src, half_t* __restrict__ dst, int n) {
  int i = blockIdx.x * blockDim.x + threadIdx.x;
  if (i < n) dst[i] = (half_t)src[i];
}

__global__ void k_pad_rows_f16(const float* __restrict__ src, half_t* __restrict__ dst,
                               int rows, int K, int rowsPad) {
  int i = blockIdx.x * blockDim.x + threadIdx.x;
  if (i >= rowsPad * K) return;
  int r = i / K;
  dst[i] = (r < rows) ? (half_t)src[i] : (half_t)0.0f;
}

__global__ void k_pad_bias(const float* __restrict__ src, float* __restrict__ dst, int n, int npad) {
  int i = blockIdx.x * blockDim.x + threadIdx.x;
  if (i < npad) dst[i] = (i < n) ? src[i] : 0.0f;
}

// Q_lvl(7x1024) = lvl_q @ Wq^T + bq   (tiny, scalar f32)
__global__ void k_qproj(const float* __restrict__ lvlq, const float* __restrict__ Wq,
                        const float* __restrict__ bq, float* __restrict__ Q) {
  int tid = blockIdx.x * blockDim.x + threadIdx.x;
  if (tid >= NLEV_ * D_) return;
  int q = tid >> 10, n = tid & (D_ - 1);
  const float* a = lvlq + q * D_;
  const float* w = Wq + (size_t)n * D_;
  float acc = 0.0f;
  for (int k = 0; k < D_; ++k) acc += a[k] * w[k];
  Q[tid] = acc + bq[n];
}

// softmax over 2 keys -> gate alpha[l][q][h] = sigma((s0-s1)/8)
__global__ void k_alpha(const float* __restrict__ Q, const float* __restrict__ K1,
                        const float* __restrict__ K2, float* __restrict__ alpha) {
  int tid = blockIdx.x * blockDim.x + threadIdx.x;
  if (tid >= HW_ * NLEV_ * NH_) return;
  int h = tid & 15;
  int q = (tid >> 4) % NLEV_;
  int l = tid / (NLEV_ * NH_);
  const float* qp = Q  + q * D_ + h * DH_;
  const float* k1 = K1 + (size_t)l * D_ + h * DH_;
  const float* k2 = K2 + (size_t)l * D_ + h * DH_;
  float s0 = 0.0f, s1 = 0.0f;
  #pragma unroll 4
  for (int j = 0; j < DH_; ++j) { s0 += qp[j] * k1[j]; s1 += qp[j] * k2[j]; }
  alpha[tid] = 1.0f / (1.0f + __expf((s1 - s0) * 0.125f));
}

// o16[(l*7+q), d] = a*V1[l,d] + (1-a)*V2[l,d]
__global__ void k_combine(const float* __restrict__ alpha, const float* __restrict__ V1,
                          const float* __restrict__ V2, half_t* __restrict__ o16) {
  int tid = blockIdx.x * blockDim.x + threadIdx.x;
  if (tid >= HW_ * NLEV_ * D_) return;
  int d = tid & (D_ - 1);
  int q = (tid >> 10) % NLEV_;
  int l = tid / (NLEV_ * D_);
  float a = alpha[(l * NLEV_ + q) * NH_ + (d >> 6)];
  float v = a * V1[(unsigned)l * D_ + d] + (1.0f - a) * V2[(unsigned)l * D_ + d];
  o16[tid] = (half_t)v;
}

// Row LayerNorm over D=1024, f32 in -> f16 out. One 256-thread block per row.
__global__ __launch_bounds__(256)
void k_ln(const float* __restrict__ X, const float* __restrict__ g,
          const float* __restrict__ b, half_t* __restrict__ Y) {
  const int row = blockIdx.x;
  const int t = threadIdx.x;
  const float4* X4 = (const float4*)(X + (size_t)row * D_);
  float4 v = X4[t];
  float s  = v.x + v.y + v.z + v.w;
  float ss = v.x * v.x + v.y * v.y + v.z * v.z + v.w * v.w;
  __shared__ float sh1[256], sh2[256];
  sh1[t] = s; sh2[t] = ss;
  __syncthreads();
  for (int off = 128; off > 0; off >>= 1) {
    if (t < off) { sh1[t] += sh1[t + off]; sh2[t] += sh2[t + off]; }
    __syncthreads();
  }
  float mean = sh1[0] * (1.0f / D_);
  float var  = sh2[0] * (1.0f / D_) - mean * mean;
  float rstd = rsqrtf(var + 1e-5f);
  half_t* Yr = Y + (size_t)row * D_;
  const float* vv = (const float*)&v;
  #pragma unroll
  for (int i = 0; i < 4; ++i) {
    int c = t * 4 + i;
    Yr[c] = (half_t)((vv[i] - mean) * rstd * g[c] + b[c]);
  }
}

// surf[v,h,w] = sh[l, v*4 + p]  (sh padded to ld=64)
__global__ void k_unpatch_surf(const float* __restrict__ sh, float* __restrict__ out) {
  int tid = blockIdx.x * blockDim.x + threadIdx.x;
  if (tid >= NVS_ * H_ * W_) return;
  int v = tid / (H_ * W_);
  int rem = tid % (H_ * W_);
  int h = rem / W_, w = rem % W_;
  int hr = h >> 1, pr = h & 1, wr = w >> 1, pc = w & 1;
  int l = hr * WR_ + wr;
  out[tid] = sh[(unsigned)l * 64 + v * 4 + pr * 2 + pc];
}

// atm[v,c,h,w] = ah[(l*7+c), v*4 + p]  (ah padded to ld=64)
__global__ void k_unpatch_atm(const float* __restrict__ ah, float* __restrict__ out) {
  int tid = blockIdx.x * blockDim.x + threadIdx.x;
  if (tid >= NVA_ * NLEV_ * H_ * W_) return;
  int v = tid / (NLEV_ * H_ * W_);
  int rem1 = tid % (NLEV_ * H_ * W_);
  int c = rem1 / (H_ * W_);
  int rem = rem1 % (H_ * W_);
  int h = rem / W_, w = rem % W_;
  int hr = h >> 1, pr = h & 1, wr = w >> 1, pc = w & 1;
  int l = hr * WR_ + wr;
  out[tid] = ah[(unsigned)(l * NLEV_ + c) * 64 + v * 4 + pr * 2 + pc];
}

// ---------------------------------------------------------------------------
extern "C" void kernel_launch(void* const* d_in, const int* in_sizes, int n_in,
                              void* d_out, int out_size, void* d_ws, size_t ws_size,
                              hipStream_t stream) {
  (void)in_sizes; (void)n_in; (void)out_size; (void)ws_size;
  const int DD = D_ * D_;

  const float* x         = (const float*)d_in[0];
  const float* lvl_q     = (const float*)d_in[1];
  const float* W_lvl_qkv = (const float*)d_in[4];
  const float* b_lvl_qkv = (const float*)d_in[5];
  const float* W_lvl_o   = (const float*)d_in[6];
  const float* b_lvl_o   = (const float*)d_in[7];
  const float* g_lvl     = (const float*)d_in[8];
  const float* be_lvl    = (const float*)d_in[9];
  const float* W_atm_qkv = (const float*)d_in[10];
  const float* b_atm_qkv = (const float*)d_in[11];
  const float* W_atm_o   = (const float*)d_in[12];
  const float* b_atm_o   = (const float*)d_in[13];
  const float* g_atm     = (const float*)d_in[14];
  const float* be_atm    = (const float*)d_in[15];
  const float* W_srf_qkv = (const float*)d_in[16];
  const float* b_srf_qkv = (const float*)d_in[17];
  const float* W_srf_o   = (const float*)d_in[18];
  const float* b_srf_o   = (const float*)d_in[19];
  const float* g_srf     = (const float*)d_in[20];
  const float* be_srf    = (const float*)d_in[21];
  const float* srf_hW    = (const float*)d_in[22];
  const float* srf_hb    = (const float*)d_in[23];
  const float* atm_hW    = (const float*)d_in[24];
  const float* atm_hb    = (const float*)d_in[25];

  float* out_surf = (float*)d_out;                       // NVS*H*W
  float* out_atm  = (float*)d_out + NVS_ * H_ * W_;      // NVA*NLEV*H*W

  // ---- workspace layout (bytes, 256-aligned, with reuse) ----
  char* ws = (char*)d_ws;
  size_t cur = 0;
  auto alloc = [&](size_t bytes) { size_t o = cur; cur = (cur + bytes + 255) & ~(size_t)255; return o; };

  half_t* w16_sv = (half_t*)(ws + alloc((size_t)DD * 2));
  half_t* w16_so = (half_t*)(ws + alloc((size_t)DD * 2));
  half_t* w16_lk = (half_t*)(ws + alloc((size_t)DD * 2));
  half_t* w16_lv = (half_t*)(ws + alloc((size_t)DD * 2));
  half_t* w16_lo = (half_t*)(ws + alloc((size_t)DD * 2));
  half_t* w16_av = (half_t*)(ws + alloc((size_t)DD * 2));
  half_t* w16_ao = (half_t*)(ws + alloc((size_t)DD * 2));
  half_t* wh_srf = (half_t*)(ws + alloc((size_t)64 * D_ * 2));
  half_t* wh_atm = (half_t*)(ws + alloc((size_t)64 * D_ * 2));
  float*  bpad_s = (float*)(ws + alloc(64 * 4));
  float*  bpad_a = (float*)(ws + alloc(64 * 4));
  float*  Qlvl   = (float*)(ws + alloc((size_t)NLEV_ * D_ * 4));
  float*  alphaB = (float*)(ws + alloc((size_t)HW_ * NLEV_ * NH_ * 4));
  half_t* x0h    = (half_t*)(ws + alloc((size_t)HW_ * D_ * 2));
  half_t* x1h    = (half_t*)(ws + alloc((size_t)HW_ * D_ * 2));
  half_t* x2h    = (half_t*)(ws + alloc((size_t)HW_ * D_ * 2));
  // K1/K2/V1/V2 f32 region (134 MB); reused later as t16 (117 MB f16)
  char*   kvreg  = ws + alloc((size_t)4 * HW_ * D_ * 4);
  float*  K1 = (float*)kvreg;
  float*  K2 = K1 + (size_t)HW_ * D_;
  float*  V1 = K2 + (size_t)HW_ * D_;
  float*  V2 = V1 + (size_t)HW_ * D_;
  half_t* t16 = (half_t*)kvreg;                          // alias (after K/V consumed)
  float*  bigf = (float*)(ws + alloc((size_t)ML_ * D_ * 4));   // t2_surf / O1 / u
  half_t* bigh = (half_t*)(ws + alloc((size_t)ML_ * D_ * 2));  // t1/y_s ; o16/z16/ya16
  float*  shh  = (float*)(ws + alloc((size_t)HW_ * 64 * 4));
  float*  ahh  = (float*)(ws + alloc((size_t)ML_ * 64 * 4));

  auto cvt = [&](const float* s, half_t* d, int n) {
    k_f32_to_f16<<<dim3((n + 255) / 256), dim3(256), 0, stream>>>(s, d, n);
  };
  // big GEMM (wave tile 64x32, block 128x128), f32 output
  auto gemm_f = [&](const half_t* A, const half_t* Bw, const float* bias, float* Cf, int M) {
    k_gemm_wmma<4, 2, 2, 4, OUT_F32><<<dim3(D_ / 128, M / 128), dim3(256), 0, stream>>>(
        A, Bw, bias, Cf, nullptr, D_, D_);
  };
  // big GEMM, f16 output
  auto gemm_h = [&](const half_t* A, const half_t* Bw, const float* bias, half_t* Ch, int M) {
    k_gemm_wmma<4, 2, 2, 4, OUT_F16><<<dim3(D_ / 128, M / 128), dim3(256), 0, stream>>>(
        A, Bw, bias, nullptr, Ch, D_, D_);
  };
  // small-N head GEMM (wave tile 32x32, block 128x64), N=64, f32 output
  auto gemm64 = [&](const half_t* A, const half_t* Bw, const float* bias, float* Cf, int M) {
    k_gemm_wmma<2, 2, 4, 2, OUT_F32><<<dim3(1, M / 128), dim3(256), 0, stream>>>(
        A, Bw, bias, Cf, nullptr, 64, D_);
  };

  // ================= surface branch (C=1 -> attn == V projection) =========
  cvt(x, x0h, HW_ * D_);
  cvt(W_srf_qkv + 2 * DD, w16_sv, DD);
  cvt(W_srf_o, w16_so, DD);
  gemm_h(x0h, w16_sv, b_srf_qkv + 2 * D_, bigh, HW_);              // t1 = x0@Wv^T+bv
  gemm_f(bigh, w16_so, b_srf_o, bigf, HW_);                        // t2 = t1@Wo^T+bo
  k_ln<<<dim3(HW_), dim3(256), 0, stream>>>(bigf, g_srf, be_srf, bigh); // y_s
  k_pad_rows_f16<<<dim3((64 * D_ + 255) / 256), dim3(256), 0, stream>>>(srf_hW, wh_srf, NVS_ * P_ * P_, D_, 64);
  k_pad_bias<<<dim3(1), dim3(64), 0, stream>>>(srf_hb, bpad_s, NVS_ * P_ * P_, 64);
  gemm64(bigh, wh_srf, bpad_s, shh, HW_);                          // head
  k_unpatch_surf<<<dim3((NVS_ * H_ * W_ + 255) / 256), dim3(256), 0, stream>>>(shh, out_surf);

  // ================= level deagg (C=2 -> sigmoid-gated V blend) ===========
  cvt(x + (size_t)HW_ * D_, x1h, HW_ * D_);
  cvt(x + (size_t)2 * HW_ * D_, x2h, HW_ * D_);
  cvt(W_lvl_qkv + DD, w16_lk, DD);        // Wk
  cvt(W_lvl_qkv + 2 * DD, w16_lv, DD);    // Wv
  cvt(W_lvl_o, w16_lo, DD);
  gemm_f(x1h, w16_lk, b_lvl_qkv + D_, K1, HW_);
  gemm_f(x2h, w16_lk, b_lvl_qkv + D_, K2, HW_);
  gemm_f(x1h, w16_lv, b_lvl_qkv + 2 * D_, V1, HW_);
  gemm_f(x2h, w16_lv, b_lvl_qkv + 2 * D_, V2, HW_);
  k_qproj<<<dim3((NLEV_ * D_ + 255) / 256), dim3(256), 0, stream>>>(lvl_q, W_lvl_qkv, b_lvl_qkv, Qlvl);
  k_alpha<<<dim3((HW_ * NLEV_ * NH_ + 255) / 256), dim3(256), 0, stream>>>(Qlvl, K1, K2, alphaB);
  k_combine<<<dim3((HW_ * NLEV_ * D_ + 255) / 256), dim3(256), 0, stream>>>(alphaB, V1, V2, bigh); // o16
  gemm_f(bigh, w16_lo, b_lvl_o, bigf, ML_);                        // O1 = o@Wo^T+bo
  k_ln<<<dim3(ML_), dim3(256), 0, stream>>>(bigf, g_lvl, be_lvl, bigh);  // z16 (overwrites o16)

  // ================= atm deagg (C=1) + head ================================
  cvt(W_atm_qkv + 2 * DD, w16_av, DD);
  cvt(W_atm_o, w16_ao, DD);
  gemm_h(bigh, w16_av, b_atm_qkv + 2 * D_, t16, ML_);              // t = z@Wv^T+bv (K/V region reused)
  gemm_f(t16, w16_ao, b_atm_o, bigf, ML_);                         // u = t@Wo^T+bo
  k_ln<<<dim3(ML_), dim3(256), 0, stream>>>(bigf, g_atm, be_atm, bigh);  // ya16 (overwrites z16)
  k_pad_rows_f16<<<dim3((64 * D_ + 255) / 256), dim3(256), 0, stream>>>(atm_hW, wh_atm, NVA_ * P_ * P_, D_, 64);
  k_pad_bias<<<dim3(1), dim3(64), 0, stream>>>(atm_hb, bpad_a, NVA_ * P_ * P_, 64);
  gemm64(bigh, wh_atm, bpad_a, ahh, ML_);                          // head
  k_unpatch_atm<<<dim3((NVA_ * NLEV_ * H_ * W_ + 255) / 256), dim3(256), 0, stream>>>(ahh, out_atm);
}
